// DynamicGNN_Memory_Module_85031762526232
// MI455X (gfx1250) — matrix operand order repaired and checked by reading, not compiled, not run on hardware
//
#include <hip/hip_runtime.h>
#include <hip/hip_bf16.h>

// ---------------------------------------------------------------------------
// DynamicGNN memory module, fused per-branch WMMA kernels for gfx1250 (MI455X)
//   B=32, N=4096, G=128, E=128, NS=64, NT=32, NE=64
// Weights resident in LDS (f16, transposed Wt[n][k]); per-wave in-place
// activation tile; v_wmma_f32_16x16x32_f16 with f32 accumulation; bias folded
// into the accumulator init; hardware v_tanh_f32 activations; next-tile
// global_prefetch of streamed inputs.
// ---------------------------------------------------------------------------

typedef __attribute__((ext_vector_type(16))) _Float16 v16h;
typedef __attribute__((ext_vector_type(8)))  _Float16 v8h;
typedef __attribute__((ext_vector_type(4)))  _Float16 v4h;
typedef __attribute__((ext_vector_type(8)))  float    v8f;
typedef __attribute__((ext_vector_type(4)))  float    v4f;

#define NNODES 4096
#define GDIM   128
#define NWAVES 8
#define THREADS 256
#define ROWS_PER_BLOCK 128          // NWAVES * 16
#define NTILES (32 * NNODES / ROWS_PER_BLOCK)   // 1024
#define GRID   512
#define XLD    256                  // activation row stride (f16 elements)

#if defined(__has_builtin)
#  if __has_builtin(__builtin_amdgcn_tanhf)
#    define FAST_TANH(x) __builtin_amdgcn_tanhf(x)
#  elif __has_builtin(__builtin_amdgcn_tanh_f32)
#    define FAST_TANH(x) __builtin_amdgcn_tanh_f32(x)
#  else
#    define FAST_TANH(x) tanhf(x)
#  endif
#else
#  define FAST_TANH(x) tanhf(x)
#endif

// ---- WMMA fragment loaders (wave32, 16x16x32 f16) -------------------------
// A 16x32: lanes 0-15 hold M=0..15 with K runs {0..7,16..23};
//          lanes 16-31 same M with K runs {8..15,24..31}.
__device__ __forceinline__ v16h ld_afrag(const _Float16* p, int hi) {
  const int s = hi << 3;
  v8h lo = *(const v8h*)(p + s);
  v8h hh = *(const v8h*)(p + 16 + s);
  v16h r;
#pragma unroll
  for (int i = 0; i < 8; ++i) { r[i] = lo[i]; r[i + 8] = hh[i]; }
  return r;
}

// B 32x16: lanes 0-15 hold N=0..15 with K runs {0..7,16..23};
//          lanes 16-31 same N with K runs {8..15,24..31} (matches A).
__device__ __forceinline__ v16h ld_bfrag(const _Float16* p, int hi) {
  const _Float16* q = p + (hi << 3);
  v8h lo = *(const v8h*)(q);
  v8h hh = *(const v8h*)(q + 16);
  v16h r;
#pragma unroll
  for (int i = 0; i < 8; ++i) { r[i] = lo[i]; r[i + 8] = hh[i]; }
  return r;
}

// One wave computes D[16][128] = X[16][K] @ W[K][128] + bias (f32 accum).
// Bias is folded into the accumulator init (C/D layout: col n = lane&15).
template<int KBLK>
__device__ __forceinline__ void wave_gemm(const _Float16* X,
                                          const _Float16* Wt,
                                          const float* bias,
                                          v8f acc[8]) {
  const int lane = threadIdx.x & 31;
  const int m    = lane & 15;
  const int hi   = lane >> 4;
  v16h a[KBLK];
#pragma unroll
  for (int kb = 0; kb < KBLK; ++kb)
    a[kb] = ld_afrag(X + m * XLD + kb * 32, hi);
#pragma unroll
  for (int nt = 0; nt < 8; ++nt) {
    const float bv = bias[nt * 16 + m];
    v8f c;
#pragma unroll
    for (int i = 0; i < 8; ++i) c[i] = bv;
    const _Float16* wrow = Wt + (size_t)(nt * 16 + m) * (KBLK * 32);
#pragma unroll
    for (int kb = 0; kb < KBLK; ++kb) {
      v16h b = ld_bfrag(wrow + kb * 32, hi);
      c = __builtin_amdgcn_wmma_f32_16x16x32_f16(
              false, a[kb], false, b, (short)0, c, false, false);
    }
    acc[nt] = c;
  }
}

// tanh(acc) -> f16 tile in LDS (in-place over X cols 0:128 is safe: all
// A-fragments were hoisted to VGPRs and same-wave LDS ops are in order).
__device__ __forceinline__ void act_store(const v8f acc[8],
                                          _Float16* Y) {
  const int lane = threadIdx.x & 31;
  const int n0 = lane & 15;
  const int hi = lane >> 4;
#pragma unroll
  for (int nt = 0; nt < 8; ++nt) {
#pragma unroll
    for (int v = 0; v < 8; ++v) {
      const int mm = v + (hi << 3);
      Y[mm * XLD + nt * 16 + n0] = (_Float16)FAST_TANH(acc[nt][v]);
    }
  }
}

// out[r][n] += tanh(acc) * mask[r]
__device__ __forceinline__ void epilogue_add(const v8f acc[8],
                                             float* out,
                                             const float* mask,
                                             int r0) {
  const int lane = threadIdx.x & 31;
  const int n0 = lane & 15;
  const int hi = lane >> 4;
#pragma unroll
  for (int v = 0; v < 8; ++v) {
    const int r = r0 + v + (hi << 3);
    const float mk = mask[r];
#pragma unroll
    for (int nt = 0; nt < 8; ++nt) {
      const size_t o = (size_t)r * GDIM + nt * 16 + n0;
      out[o] += FAST_TANH(acc[nt][v]) * mk;
    }
  }
}

// Cooperative: global W[K][128] f32 -> LDS Wt[128][K] f16, 4 k's per store.
__device__ __forceinline__ void loadWT(const float* W, _Float16* Wt,
                                       int K, int tid) {
  const int total = (K >> 2) * 128;
  for (int idx = tid; idx < total; idx += THREADS) {
    const int kq = idx >> 7;
    const int n  = idx & 127;
    const int k  = kq << 2;
    v4h h;
    h[0] = (_Float16)W[(size_t)(k + 0) * 128 + n];
    h[1] = (_Float16)W[(size_t)(k + 1) * 128 + n];
    h[2] = (_Float16)W[(size_t)(k + 2) * 128 + n];
    h[3] = (_Float16)W[(size_t)(k + 3) * 128 + n];
    *(v4h*)(Wt + n * K + k) = h;
  }
}

__device__ __forceinline__ void loadVec(const float* s, float* d, int n, int tid) {
  for (int i = tid; i < n; i += THREADS) d[i] = s[i];
}

// f32 global rows -> f16 LDS tile columns [c0, c0+C) for 16 rows.
__device__ __forceinline__ void fill_rows(const float* src, int src_ld,
                                          _Float16* X, int c0, int C,
                                          int r0, int lane) {
  const int quads_per_row = C >> 2;
  const int total = 16 * quads_per_row;
  for (int idx = lane; idx < total; idx += 32) {
    const int mm = idx / quads_per_row;
    const int k4 = (idx % quads_per_row) << 2;
    v4f v = *(const v4f*)(src + (size_t)(r0 + mm) * src_ld + k4);
    v4h h;
#pragma unroll
    for (int i = 0; i < 4; ++i) h[i] = (_Float16)v[i];
    *(v4h*)(X + mm * XLD + c0 + k4) = h;
  }
}

// Prefetch next tile's 16-row f32 slab (one address per lane; HW pulls the
// whole cacheline -> global_prefetch_b8 on gfx1250).
__device__ __forceinline__ void prefetch_rows(const float* src, int src_ld,
                                              int r0, int lane) {
  const int mm = lane & 15;
  const int off = (lane >> 4) * (src_ld >> 1);   // two probes per row
  __builtin_prefetch(src + (size_t)(r0 + mm) * src_ld + off, 0, 1);
}

// ---------------------------------------------------------------------------
// State branch: se = tanh(sidx@s2e); gi = mlp2(se, s2h); sm = mlp2(gi||hx, smc)
// out = hx + sm * smask   (full overwrite initializes d_out)
// ---------------------------------------------------------------------------
__global__ __launch_bounds__(THREADS) void gnn_state_kernel(
    const float* __restrict__ hx, const float* __restrict__ sidx,
    const float* __restrict__ smask,
    const float* __restrict__ s2e_W,  const float* __restrict__ s2e_b,
    const float* __restrict__ s2h_W1, const float* __restrict__ s2h_b1,
    const float* __restrict__ s2h_W2, const float* __restrict__ s2h_b2,
    const float* __restrict__ smc_W1, const float* __restrict__ smc_b1,
    const float* __restrict__ smc_W2, const float* __restrict__ smc_b2,
    float* __restrict__ out) {
  extern __shared__ char smem[];
  _Float16* w_s2e  = (_Float16*)smem;          // 128 x 64
  _Float16* w_s2h1 = w_s2e  + 128 * 64;        // 128 x 128
  _Float16* w_s2h2 = w_s2h1 + 128 * 128;       // 128 x 128
  _Float16* w_smc1 = w_s2h2 + 128 * 128;       // 128 x 256
  _Float16* w_smc2 = w_smc1 + 128 * 256;       // 128 x 128
  float* b_s2e  = (float*)(w_smc2 + 128 * 128);
  float* b_s2h1 = b_s2e  + 128;
  float* b_s2h2 = b_s2h1 + 128;
  float* b_smc1 = b_s2h2 + 128;
  float* b_smc2 = b_smc1 + 128;
  _Float16* acts = (_Float16*)(b_smc2 + 128);  // NWAVES * 16 * XLD

  const int tid = threadIdx.x;
  loadWT(s2e_W,  w_s2e,  64,  tid);
  loadWT(s2h_W1, w_s2h1, 128, tid);
  loadWT(s2h_W2, w_s2h2, 128, tid);
  loadWT(smc_W1, w_smc1, 256, tid);
  loadWT(smc_W2, w_smc2, 128, tid);
  loadVec(s2e_b,  b_s2e,  128, tid);
  loadVec(s2h_b1, b_s2h1, 128, tid);
  loadVec(s2h_b2, b_s2h2, 128, tid);
  loadVec(smc_b1, b_smc1, 128, tid);
  loadVec(smc_b2, b_smc2, 128, tid);
  __syncthreads();

  const int wave = tid >> 5;
  const int lane = tid & 31;
  _Float16* X = acts + wave * 16 * XLD;

  for (int tile = blockIdx.x; tile < NTILES; tile += gridDim.x) {
    const int r0 = tile * ROWS_PER_BLOCK + wave * 16;
    const int tn = tile + gridDim.x;
    if (tn < NTILES) {               // prefetch next tile's streamed inputs
      const int rn = tn * ROWS_PER_BLOCK + wave * 16;
      prefetch_rows(sidx, 64, rn, lane);
      prefetch_rows(hx, GDIM, rn, lane);
    }
    fill_rows(sidx, 64, X, 0, 64, r0, lane);

    v8f acc[8];
    wave_gemm<2>(X, w_s2e,  b_s2e,  acc); act_store(acc, X);  // se
    wave_gemm<4>(X, w_s2h1, b_s2h1, acc); act_store(acc, X);
    wave_gemm<4>(X, w_s2h2, b_s2h2, acc); act_store(acc, X);  // gi
    fill_rows(hx, GDIM, X, 128, 128, r0, lane);
    wave_gemm<8>(X, w_smc1, b_smc1, acc); act_store(acc, X);
    wave_gemm<4>(X, w_smc2, b_smc2, acc);

    // out = hx + tanh(acc) * mask
    const int n0 = lane & 15, hi = lane >> 4;
#pragma unroll
    for (int v = 0; v < 8; ++v) {
      const int r = r0 + v + (hi << 3);
      const float mk = smask[r];
#pragma unroll
      for (int nt = 0; nt < 8; ++nt) {
        const size_t o = (size_t)r * GDIM + nt * 16 + n0;
        out[o] = hx[o] + FAST_TANH(acc[nt][v]) * mk;
      }
    }
  }
}

// ---------------------------------------------------------------------------
// Edge branch: gi = mlp2(edge_emb[type] || hx[gather], e2h);
//              em = mlp2(gi || hx, emc); out += em * emask
// ---------------------------------------------------------------------------
__global__ __launch_bounds__(THREADS) void gnn_edge_kernel(
    const float* __restrict__ hx, const int* __restrict__ eidx,
    const int* __restrict__ etyp, const float* __restrict__ emask,
    const float* __restrict__ edge_emb,
    const float* __restrict__ e2h_W1, const float* __restrict__ e2h_b1,
    const float* __restrict__ e2h_W2, const float* __restrict__ e2h_b2,
    const float* __restrict__ emc_W1, const float* __restrict__ emc_b1,
    const float* __restrict__ emc_W2, const float* __restrict__ emc_b2,
    float* __restrict__ out) {
  extern __shared__ char smem[];
  _Float16* w_e2h1 = (_Float16*)smem;          // 128 x 256
  _Float16* w_e2h2 = w_e2h1 + 128 * 256;       // 128 x 128
  _Float16* w_emc1 = w_e2h2 + 128 * 128;       // 128 x 256
  _Float16* w_emc2 = w_emc1 + 128 * 256;       // 128 x 128
  _Float16* emb16  = w_emc2 + 128 * 128;       // 64 x 128 row-major
  float* b_e2h1 = (float*)(emb16 + 64 * 128);
  float* b_e2h2 = b_e2h1 + 128;
  float* b_emc1 = b_e2h2 + 128;
  float* b_emc2 = b_emc1 + 128;
  _Float16* acts = (_Float16*)(b_emc2 + 128);

  const int tid = threadIdx.x;
  loadWT(e2h_W1, w_e2h1, 256, tid);
  loadWT(e2h_W2, w_e2h2, 128, tid);
  loadWT(emc_W1, w_emc1, 256, tid);
  loadWT(emc_W2, w_emc2, 128, tid);
  for (int i = tid; i < 64 * 128 / 4; i += THREADS) {
    v4f v = *(const v4f*)(edge_emb + i * 4);
    v4h h;
#pragma unroll
    for (int j = 0; j < 4; ++j) h[j] = (_Float16)v[j];
    *(v4h*)(emb16 + i * 4) = h;
  }
  loadVec(e2h_b1, b_e2h1, 128, tid);
  loadVec(e2h_b2, b_e2h2, 128, tid);
  loadVec(emc_b1, b_emc1, 128, tid);
  loadVec(emc_b2, b_emc2, 128, tid);
  __syncthreads();

  const int wave = tid >> 5;
  const int lane = tid & 31;
  _Float16* X = acts + wave * 16 * XLD;

  for (int tile = blockIdx.x; tile < NTILES; tile += gridDim.x) {
    const int r0 = tile * ROWS_PER_BLOCK + wave * 16;
    const size_t bq = (size_t)(r0 >> 12) * NNODES;   // batch base row
    const int tn = tile + gridDim.x;
    if (tn < NTILES) {
      const int rn = tn * ROWS_PER_BLOCK + wave * 16;
      prefetch_rows(hx, GDIM, rn, lane);
      if (lane < 16) __builtin_prefetch(eidx + rn + lane, 0, 1);
    }

    // X = [ edge_emb[type] || hx[b, edge_idx] ]
    for (int idx = lane; idx < 16 * 32; idx += 32) {
      const int mm = idx >> 5, k4 = (idx & 31) << 2;
      const int r = r0 + mm;
      const int et = etyp[r];
      const int ng = eidx[r];
      *(v4h*)(X + mm * XLD + k4) = *(const v4h*)(emb16 + et * 128 + k4);
      v4f o = *(const v4f*)(hx + (bq + ng) * GDIM + k4);
      v4h h;
#pragma unroll
      for (int i = 0; i < 4; ++i) h[i] = (_Float16)o[i];
      *(v4h*)(X + mm * XLD + 128 + k4) = h;
    }

    v8f acc[8];
    wave_gemm<8>(X, w_e2h1, b_e2h1, acc); act_store(acc, X);
    wave_gemm<4>(X, w_e2h2, b_e2h2, acc); act_store(acc, X);  // gi
    fill_rows(hx, GDIM, X, 128, 128, r0, lane);
    wave_gemm<8>(X, w_emc1, b_emc1, acc); act_store(acc, X);
    wave_gemm<4>(X, w_emc2, b_emc2, acc);
    epilogue_add(acc, out, emask, r0);
  }
}

// ---------------------------------------------------------------------------
// Interaction branch: gi = mlp2(action_emb || touch, a2h);
//                     im = mlp2(gi || hx, imc); out += im * tmask
// ---------------------------------------------------------------------------
__global__ __launch_bounds__(THREADS) void gnn_inter_kernel(
    const float* __restrict__ hx, const float* __restrict__ ae,
    const float* __restrict__ touch, const float* __restrict__ tmask,
    const float* __restrict__ a2h_W1, const float* __restrict__ a2h_b1,
    const float* __restrict__ a2h_W2, const float* __restrict__ a2h_b2,
    const float* __restrict__ imc_W1, const float* __restrict__ imc_b1,
    const float* __restrict__ imc_W2, const float* __restrict__ imc_b2,
    float* __restrict__ out) {
  extern __shared__ char smem[];
  _Float16* w_a2h1 = (_Float16*)smem;          // 128 x 160
  _Float16* w_a2h2 = w_a2h1 + 128 * 160;       // 128 x 128
  _Float16* w_imc1 = w_a2h2 + 128 * 128;       // 128 x 256
  _Float16* w_imc2 = w_imc1 + 128 * 256;       // 128 x 128
  float* b_a2h1 = (float*)(w_imc2 + 128 * 128);
  float* b_a2h2 = b_a2h1 + 128;
  float* b_imc1 = b_a2h2 + 128;
  float* b_imc2 = b_imc1 + 128;
  _Float16* acts = (_Float16*)(b_imc2 + 128);

  const int tid = threadIdx.x;
  loadWT(a2h_W1, w_a2h1, 160, tid);
  loadWT(a2h_W2, w_a2h2, 128, tid);
  loadWT(imc_W1, w_imc1, 256, tid);
  loadWT(imc_W2, w_imc2, 128, tid);
  loadVec(a2h_b1, b_a2h1, 128, tid);
  loadVec(a2h_b2, b_a2h2, 128, tid);
  loadVec(imc_b1, b_imc1, 128, tid);
  loadVec(imc_b2, b_imc2, 128, tid);
  __syncthreads();

  const int wave = tid >> 5;
  const int lane = tid & 31;
  _Float16* X = acts + wave * 16 * XLD;

  for (int tile = blockIdx.x; tile < NTILES; tile += gridDim.x) {
    const int r0 = tile * ROWS_PER_BLOCK + wave * 16;
    const int bb = r0 >> 12;
    const int tn = tile + gridDim.x;
    if (tn < NTILES) {
      const int rn = tn * ROWS_PER_BLOCK + wave * 16;
      prefetch_rows(touch, 32, rn, lane);
      prefetch_rows(hx, GDIM, rn, lane);
    }

    // X = [ action_emb (broadcast, L1-hot) || touch ],  K = 160
    for (int idx = lane; idx < 16 * 32; idx += 32) {
      const int mm = idx >> 5, k4 = (idx & 31) << 2;
      v4f v = *(const v4f*)(ae + (size_t)bb * 128 + k4);
      v4h h;
#pragma unroll
      for (int i = 0; i < 4; ++i) h[i] = (_Float16)v[i];
      *(v4h*)(X + mm * XLD + k4) = h;
    }
    fill_rows(touch, 32, X, 128, 32, r0, lane);

    v8f acc[8];
    wave_gemm<5>(X, w_a2h1, b_a2h1, acc); act_store(acc, X);
    wave_gemm<4>(X, w_a2h2, b_a2h2, acc); act_store(acc, X);  // gi
    fill_rows(hx, GDIM, X, 128, 128, r0, lane);
    wave_gemm<8>(X, w_imc1, b_imc1, acc); act_store(acc, X);
    wave_gemm<4>(X, w_imc2, b_imc2, acc);
    epilogue_add(acc, out, tmask, r0);
  }
}

// ---------------------------------------------------------------------------
extern "C" void kernel_launch(void* const* d_in, const int* in_sizes, int n_in,
                              void* d_out, int out_size, void* d_ws, size_t ws_size,
                              hipStream_t stream) {
  (void)in_sizes; (void)n_in; (void)d_ws; (void)ws_size; (void)out_size;
  const float* hx    = (const float*)d_in[0];
  const float* ae    = (const float*)d_in[1];
  const float* touch = (const float*)d_in[2];
  const float* tmask = (const float*)d_in[3];
  const float* sidx  = (const float*)d_in[4];
  // d_in[5] = batch_state_legal (unused by reference)
  const float* smask = (const float*)d_in[6];
  const int*   eidx  = (const int*)d_in[7];
  const int*   etyp  = (const int*)d_in[8];
  const float* emask = (const float*)d_in[9];
  const float* P[27];
  for (int i = 0; i < 27; ++i) P[i] = (const float*)d_in[10 + i];
  // P: 0 s2e_W 1 s2e_b 2 s2h_W1 3 s2h_b1 4 s2h_W2 5 s2h_b2
  //    6 smc_W1 7 smc_b1 8 smc_W2 9 smc_b2 10 edge_emb
  //    11 e2h_W1 12 e2h_b1 13 e2h_W2 14 e2h_b2 15 emc_W1 16 emc_b1 17 emc_W2 18 emc_b2
  //    19 a2h_W1 20 a2h_b1 21 a2h_W2 22 a2h_b2 23 imc_W1 24 imc_b1 25 imc_W2 26 imc_b2
  float* out = (float*)d_out;

  dim3 grid(GRID), block(THREADS);
  const size_t act_bytes = (size_t)NWAVES * 16 * XLD * 2;
  const size_t lds_state = (size_t)(128*64 + 3*128*128 + 128*256) * 2 + 5*128*4 + act_bytes;
  const size_t lds_edge  = (size_t)(2*128*256 + 2*128*128 + 64*128) * 2 + 4*128*4 + act_bytes;
  const size_t lds_inter = (size_t)(128*160 + 2*128*128 + 128*256) * 2 + 4*128*4 + act_bytes;

  gnn_state_kernel<<<grid, block, lds_state, stream>>>(
      hx, sidx, smask,
      P[0], P[1], P[2], P[3], P[4], P[5], P[6], P[7], P[8], P[9], out);
  gnn_edge_kernel<<<grid, block, lds_edge, stream>>>(
      hx, eidx, etyp, emask, P[10],
      P[11], P[12], P[13], P[14], P[15], P[16], P[17], P[18], out);
  gnn_inter_kernel<<<grid, block, lds_inter, stream>>>(
      hx, ae, touch, tmask,
      P[19], P[20], P[21], P[22], P[23], P[24], P[25], P[26], out);
}
